// LlamaSkipMLP_60232621359626
// MI455X (gfx1250) — matrix operand order repaired and checked by reading, not compiled, not run on hardware
//
#include <hip/hip_runtime.h>
#include <stdint.h>

// ---------------- problem constants ----------------
#define HIDDEN   4096
#define INTER    11008
#define K_ACTIVE 3302
#define TOKENS   8192
#define KPAD     3328        // K_ACTIVE rounded up to a multiple of 64

// ---------------- tiling ----------------
#define TM 128               // block tile: tokens
#define TN 64                // block tile: N (active neurons / hidden cols)
#define KC 32                // K chunk per WMMA (bf16 16x16x32)
#define NKCH (HIDDEN / KC)   // 128 chunks, stage 1
#define NKCD (KPAD / KC)     // 104 chunks, stage 2

typedef __attribute__((ext_vector_type(16))) __bf16 v16bf;
typedef __attribute__((ext_vector_type(8)))  float  v8f;

// ---------------- f32 -> bf16 (RNE) ----------------
__device__ __forceinline__ uint16_t f2bf(float f) {
  union { float f; uint32_t u; } c; c.f = f;
  uint32_t u = c.u;
  u += 0x7FFFu + ((u >> 16) & 1u);
  return (uint16_t)(u >> 16);
}
__device__ __forceinline__ uint32_t pk2(float a, float b) {
  return (uint32_t)f2bf(a) | ((uint32_t)f2bf(b) << 16);
}
__device__ __forceinline__ uint4 cvt8(float4 a, float4 b) {
  uint4 r;
  r.x = pk2(a.x, a.y); r.y = pk2(a.z, a.w);
  r.z = pk2(b.x, b.y); r.w = pk2(b.z, b.w);
  return r;
}

// ---------------- CDNA5 async global->LDS copy (fallback path) ----------------
__device__ __forceinline__ void async_b128(const void* gsrc, uint32_t lds_off) {
  asm volatile("global_load_async_to_lds_b128 %0, %1, off"
               :
               : "v"(lds_off), "v"(gsrc)
               : "memory");
}
#if __has_builtin(__builtin_amdgcn_s_wait_asynccnt)
#define WAIT_ASYNC(n) __builtin_amdgcn_s_wait_asynccnt(n)
#else
#define WAIT_ASYNC(n) asm volatile("s_wait_asynccnt %0" :: "i"(n) : "memory")
#endif

// ---------------- CDNA5 Tensor Data Mover (preferred path) ----------------
// 2D tile DMA: one wave issues one instruction per tile; descriptor per
// CDNA5 ISA (count=1, type=2, data_size=2B).
#if __has_builtin(__builtin_amdgcn_tensor_load_to_lds) && \
    __has_builtin(__builtin_amdgcn_s_wait_tensorcnt)
#define HAVE_TDM 1
typedef uint32_t tdm_v4u __attribute__((ext_vector_type(4)));
typedef int      tdm_v8i __attribute__((ext_vector_type(8)));
typedef int      tdm_v4i __attribute__((ext_vector_type(4)));

__device__ __forceinline__ void tdm_load_2d(uint32_t lds_addr, const void* gsrc,
                                            uint32_t tile_rows, uint32_t stride_elems) {
  const uint64_t ga = (uint64_t)(uintptr_t)gsrc;
  tdm_v4u g0;
  g0[0] = 1u;                                   // count=1 (valid), no gather
  g0[1] = lds_addr;                             // lds_addr[31:0]
  g0[2] = (uint32_t)ga;                         // global_addr[31:0] (tile start)
  g0[3] = (uint32_t)(ga >> 32) | (2u << 30);    // global_addr[56:32] | type=2
  const uint32_t td0 = stride_elems;            // tensor_dim0 (tiles never OOB)
  const uint32_t td1 = 1u << 20;                // tensor_dim1 (large, no OOB)
  tdm_v8i g1;
  g1[0] = (int)(1u << 16);                      // workgroup_mask=0 | data_size=1 (2B)
  g1[1] = (int)((td0 & 0xFFFFu) << 16);         // abar_addr=0 | tensor_dim0 lo16
  g1[2] = (int)((td0 >> 16) | ((td1 & 0xFFFFu) << 16)); // dim0 hi16 | dim1 lo16
  g1[3] = (int)((td1 >> 16) | ((uint32_t)KC << 16));    // dim1 hi16 | tile_dim0=32
  g1[4] = (int)(tile_rows & 0xFFFFu);           // tile_dim1 | tile_dim2=0
  g1[5] = (int)stride_elems;                    // tensor_dim0_stride lo32
  g1[6] = 0;                                    // dim0_stride hi16 | dim1_stride lo16
  g1[7] = 0;                                    // tensor_dim1_stride hi32
  const tdm_v4i gz = {0, 0, 0, 0};
#if __clang_major__ >= 23
  const tdm_v8i gz8 = {0, 0, 0, 0, 0, 0, 0, 0};
  __builtin_amdgcn_tensor_load_to_lds(g0, g1, gz, gz, gz8, 0);
#else
  __builtin_amdgcn_tensor_load_to_lds(g0, g1, gz, gz, 0);
#endif
}
#define WAIT_TENSOR(n) __builtin_amdgcn_s_wait_tensorcnt(n)
#endif

// ---------------- WMMA fragment helpers ----------------
// A-frag (16x32): elems 0..7 = K kb..kb+7, elems 8..15 = K kb+16..kb+23 -> hi = lo+16
// B-frag (32x16): elems 0..7 = K kb..kb+7, elems 8..15 = K kb+16..kb+23 -> hi = lo+16
__device__ __forceinline__ v16bf load_frag(const uint16_t* lo, const uint16_t* hi) {
  union { uint4 u[2]; v16bf v; } cv;
  cv.u[0] = *(const uint4*)lo;
  cv.u[1] = *(const uint4*)hi;
  return cv.v;
}
__device__ __forceinline__ v8f wmma_bf16(v16bf a, v16bf b, v8f c) {
  return __builtin_amdgcn_wmma_f32_16x16x32_bf16(false, a, false, b, (short)0, c,
                                                 false, false);
}

// ============================================================
// Precompute kernels (once per call; ~250 MB @ 23.3 TB/s ~ 11 us)
// ============================================================
__global__ __launch_bounds__(256)
void cvt_x_kernel(const float* __restrict__ x, uint16_t* __restrict__ xb) {
  size_t e = ((size_t)blockIdx.x * 256 + threadIdx.x) * 8;
  float4 a = *(const float4*)(x + e);
  float4 b = *(const float4*)(x + e + 4);
  *(uint4*)(xb + e) = cvt8(a, b);
}

__global__ __launch_bounds__(256)
void gather_gu_kernel(const float* __restrict__ Wg, const float* __restrict__ Wu,
                      const int* __restrict__ idx,
                      uint16_t* __restrict__ wg, uint16_t* __restrict__ wu) {
  int r  = blockIdx.x;
  int c  = (blockIdx.y * 256 + threadIdx.x) * 8;
  int kc = r < K_ACTIVE ? r : (K_ACTIVE - 1);
  size_t src = (size_t)idx[kc] * HIDDEN + c;
  size_t dst = (size_t)r * HIDDEN + c;
  float4 a = *(const float4*)(Wg + src);
  float4 b = *(const float4*)(Wg + src + 4);
  *(uint4*)(wg + dst) = cvt8(a, b);
  a = *(const float4*)(Wu + src);
  b = *(const float4*)(Wu + src + 4);
  *(uint4*)(wu + dst) = cvt8(a, b);
}

__global__ __launch_bounds__(256)
void gather_wd_kernel(const float* __restrict__ Wd, const int* __restrict__ idx,
                      uint16_t* __restrict__ wd) {
  int n  = blockIdx.x;
  int k0 = (blockIdx.y * 256 + threadIdx.x) * 8;
  if (k0 >= KPAD) return;
  float v[8];
#pragma unroll
  for (int j = 0; j < 8; ++j) {
    int kk  = k0 + j;
    int col = idx[kk < K_ACTIVE ? kk : (K_ACTIVE - 1)];
    v[j] = Wd[(size_t)n * INTER + col];
  }
  uint4 r;
  r.x = pk2(v[0], v[1]); r.y = pk2(v[2], v[3]);
  r.z = pk2(v[4], v[5]); r.w = pk2(v[6], v[7]);
  *(uint4*)(wd + (size_t)n * KPAD + k0) = r;
}

// ============================================================
// Kernel 1: h = silu(x·Wg'^T) * (x·Wu'^T)   (bf16 out, TDM/async double-buffer)
// ============================================================
__global__ __launch_bounds__(256)
void gateup_kernel(const uint16_t* __restrict__ xb, const uint16_t* __restrict__ wg,
                   const uint16_t* __restrict__ wu, uint16_t* __restrict__ hbuf) {
  __shared__ __align__(16) uint16_t sX[2][TM * KC];   // 2 x 8 KB
  __shared__ __align__(16) uint16_t sG[2][TN * KC];   // 2 x 4 KB
  __shared__ __align__(16) uint16_t sU[2][TN * KC];   // 2 x 4 KB

  const int tid  = threadIdx.x;
  const int lane = tid & 31;
  const int wM   = (tid >> 5) >> 1;   // 0..3
  const int wN   = (tid >> 5) & 1;    // 0..1
  const int tRow0 = blockIdx.x * TM;
  const int nCol0 = blockIdx.y * TN;

#ifdef HAVE_TDM
  const bool leader = (tid < 32);     // one wave drives the TDM
  auto issue = [&](int h0, int b) {
    tdm_load_2d((uint32_t)(uintptr_t)&sX[b][0],
                xb + (size_t)tRow0 * HIDDEN + h0, TM, HIDDEN);
    tdm_load_2d((uint32_t)(uintptr_t)&sG[b][0],
                wg + (size_t)nCol0 * HIDDEN + h0, TN, HIDDEN);
    tdm_load_2d((uint32_t)(uintptr_t)&sU[b][0],
                wu + (size_t)nCol0 * HIDDEN + h0, TN, HIDDEN);
  };
#else
  const int eA0 = tid * 8,         rA0 = eA0 >> 5, cA0 = eA0 & 31;
  const int eA1 = (tid + 256) * 8, rA1 = eA1 >> 5, cA1 = eA1 & 31;
  const int eB  = tid * 8,         rB  = eB  >> 5, cB  = eB  & 31;
  auto issue = [&](int h0, int b) {
    async_b128(xb + (size_t)(tRow0 + rA0) * HIDDEN + h0 + cA0,
               (uint32_t)(uintptr_t)&sX[b][rA0 * KC + cA0]);
    async_b128(xb + (size_t)(tRow0 + rA1) * HIDDEN + h0 + cA1,
               (uint32_t)(uintptr_t)&sX[b][rA1 * KC + cA1]);
    async_b128(wg + (size_t)(nCol0 + rB) * HIDDEN + h0 + cB,
               (uint32_t)(uintptr_t)&sG[b][rB * KC + cB]);
    async_b128(wu + (size_t)(nCol0 + rB) * HIDDEN + h0 + cB,
               (uint32_t)(uintptr_t)&sU[b][rB * KC + cB]);
  };
#endif

  const v8f vz = {0.f,0.f,0.f,0.f,0.f,0.f,0.f,0.f};
  v8f accG[2][2], accU[2][2];
#pragma unroll
  for (int i = 0; i < 2; ++i)
#pragma unroll
    for (int j = 0; j < 2; ++j) { accG[i][j] = vz; accU[i][j] = vz; }

  const int mr  = lane & 15;
  const int akb = (lane >> 4) << 3;   // 0 / 8
  const int bkb = (lane >> 4) << 3;   // 0 / 8

#ifdef HAVE_TDM
  if (leader) issue(0, 0);
#else
  issue(0, 0);
#endif
  for (int k = 0; k < NKCH; ++k) {
    const int cur = k & 1;
#ifdef HAVE_TDM
    if (leader) {
      if (k + 1 < NKCH) { issue((k + 1) * KC, cur ^ 1); WAIT_TENSOR(3); }
      else              { WAIT_TENSOR(0); }
    }
#else
    if (k + 1 < NKCH) { issue((k + 1) * KC, cur ^ 1); WAIT_ASYNC(4); }
    else              { WAIT_ASYNC(0); }
#endif
    __syncthreads();   // chunk k visible to all waves

    v16bf aF[2], gF[2], uF[2];
#pragma unroll
    for (int mt = 0; mt < 2; ++mt) {
      const uint16_t* p = &sX[cur][(wM*32 + mt*16 + mr) * KC + akb];
      aF[mt] = load_frag(p, p + 16);
    }
#pragma unroll
    for (int nt = 0; nt < 2; ++nt) {
      const uint16_t* pg = &sG[cur][(wN*32 + nt*16 + mr) * KC + bkb];
      gF[nt] = load_frag(pg, pg + 16);
      const uint16_t* pu = &sU[cur][(wN*32 + nt*16 + mr) * KC + bkb];
      uF[nt] = load_frag(pu, pu + 16);
    }
#pragma unroll
    for (int mt = 0; mt < 2; ++mt)
#pragma unroll
      for (int nt = 0; nt < 2; ++nt) {
        accG[mt][nt] = wmma_bf16(aF[mt], gF[nt], accG[mt][nt]);
        accU[mt][nt] = wmma_bf16(aF[mt], uF[nt], accU[mt][nt]);
      }
    __syncthreads();   // all waves done reading buf[cur] before refill
  }

  const int mOff = (lane >> 4) << 3;
  const int nOff = lane & 15;
#pragma unroll
  for (int mt = 0; mt < 2; ++mt)
#pragma unroll
    for (int nt = 0; nt < 2; ++nt)
#pragma unroll
      for (int r = 0; r < 8; ++r) {
        float g = accG[mt][nt][r];
        float u = accU[mt][nt][r];
        float s = g / (1.0f + __expf(-g));
        float hv = s * u;
        int row = tRow0 + wM*32 + mt*16 + mOff + r;
        int col = nCol0 + wN*32 + nt*16 + nOff;
        hbuf[(size_t)row * KPAD + col] = (col < K_ACTIVE) ? f2bf(hv) : (uint16_t)0;
      }
}

// ============================================================
// Kernel 2: out = h · Wd'^T   (f32 out, TDM/async double-buffer)
// ============================================================
__global__ __launch_bounds__(256)
void down_kernel(const uint16_t* __restrict__ hbuf, const uint16_t* __restrict__ wd,
                 float* __restrict__ out) {
  __shared__ __align__(16) uint16_t sH[2][TM * KC];   // 2 x 8 KB
  __shared__ __align__(16) uint16_t sW[2][TN * KC];   // 2 x 4 KB

  const int tid  = threadIdx.x;
  const int lane = tid & 31;
  const int wM   = (tid >> 5) >> 1;
  const int wN   = (tid >> 5) & 1;
  const int tRow0 = blockIdx.x * TM;
  const int hCol0 = blockIdx.y * TN;

#ifdef HAVE_TDM
  const bool leader = (tid < 32);
  auto issue = [&](int k0, int b) {
    tdm_load_2d((uint32_t)(uintptr_t)&sH[b][0],
                hbuf + (size_t)tRow0 * KPAD + k0, TM, KPAD);
    tdm_load_2d((uint32_t)(uintptr_t)&sW[b][0],
                wd + (size_t)hCol0 * KPAD + k0, TN, KPAD);
  };
#else
  const int eA0 = tid * 8,         rA0 = eA0 >> 5, cA0 = eA0 & 31;
  const int eA1 = (tid + 256) * 8, rA1 = eA1 >> 5, cA1 = eA1 & 31;
  const int eB  = tid * 8,         rB  = eB  >> 5, cB  = eB  & 31;
  auto issue = [&](int k0, int b) {
    async_b128(hbuf + (size_t)(tRow0 + rA0) * KPAD + k0 + cA0,
               (uint32_t)(uintptr_t)&sH[b][rA0 * KC + cA0]);
    async_b128(hbuf + (size_t)(tRow0 + rA1) * KPAD + k0 + cA1,
               (uint32_t)(uintptr_t)&sH[b][rA1 * KC + cA1]);
    async_b128(wd + (size_t)(hCol0 + rB) * KPAD + k0 + cB,
               (uint32_t)(uintptr_t)&sW[b][rB * KC + cB]);
  };
#endif

  const v8f vz = {0.f,0.f,0.f,0.f,0.f,0.f,0.f,0.f};
  v8f acc[2][2];
#pragma unroll
  for (int i = 0; i < 2; ++i)
#pragma unroll
    for (int j = 0; j < 2; ++j) acc[i][j] = vz;

  const int mr  = lane & 15;
  const int akb = (lane >> 4) << 3;
  const int bkb = (lane >> 4) << 3;

#ifdef HAVE_TDM
  if (leader) issue(0, 0);
#else
  issue(0, 0);
#endif
  for (int k = 0; k < NKCD; ++k) {
    const int cur = k & 1;
#ifdef HAVE_TDM
    if (leader) {
      if (k + 1 < NKCD) { issue((k + 1) * KC, cur ^ 1); WAIT_TENSOR(2); }
      else              { WAIT_TENSOR(0); }
    }
#else
    if (k + 1 < NKCD) { issue((k + 1) * KC, cur ^ 1); WAIT_ASYNC(3); }
    else              { WAIT_ASYNC(0); }
#endif
    __syncthreads();

    v16bf aF[2], bF[2];
#pragma unroll
    for (int mt = 0; mt < 2; ++mt) {
      const uint16_t* p = &sH[cur][(wM*32 + mt*16 + mr) * KC + akb];
      aF[mt] = load_frag(p, p + 16);
    }
#pragma unroll
    for (int nt = 0; nt < 2; ++nt) {
      const uint16_t* p = &sW[cur][(wN*32 + nt*16 + mr) * KC + bkb];
      bF[nt] = load_frag(p, p + 16);
    }
#pragma unroll
    for (int mt = 0; mt < 2; ++mt)
#pragma unroll
      for (int nt = 0; nt < 2; ++nt)
        acc[mt][nt] = wmma_bf16(aF[mt], bF[nt], acc[mt][nt]);
    __syncthreads();
  }

  const int mOff = (lane >> 4) << 3;
  const int nOff = lane & 15;
#pragma unroll
  for (int mt = 0; mt < 2; ++mt)
#pragma unroll
    for (int nt = 0; nt < 2; ++nt)
#pragma unroll
      for (int r = 0; r < 8; ++r) {
        int row = tRow0 + wM*32 + mt*16 + mOff + r;
        int col = hCol0 + wN*32 + nt*16 + nOff;
        out[(size_t)row * HIDDEN + col] = acc[mt][nt][r];
      }
}

// ============================================================
extern "C" void kernel_launch(void* const* d_in, const int* in_sizes, int n_in,
                              void* d_out, int out_size, void* d_ws, size_t ws_size,
                              hipStream_t stream) {
  (void)in_sizes; (void)n_in; (void)out_size; (void)ws_size;
  const float* x  = (const float*)d_in[0];
  const float* Wg = (const float*)d_in[1];
  const float* Wu = (const float*)d_in[2];
  const float* Wd = (const float*)d_in[3];
  const int*   id = (const int*)  d_in[4];
  float*       out = (float*)d_out;

  // workspace layout (bf16): xb | wg' | wu' | wd' | h   (~194 MB total)
  uint16_t* xb = (uint16_t*)d_ws;
  uint16_t* wg = xb + (size_t)TOKENS * HIDDEN;
  uint16_t* wu = wg + (size_t)KPAD * HIDDEN;
  uint16_t* wd = wu + (size_t)KPAD * HIDDEN;
  uint16_t* hb = wd + (size_t)HIDDEN * KPAD;

  dim3 blk(256);
  cvt_x_kernel    <<<dim3((TOKENS * HIDDEN) / 2048),     blk, 0, stream>>>(x, xb);
  gather_gu_kernel<<<dim3(KPAD, HIDDEN / 2048),          blk, 0, stream>>>(Wg, Wu, id, wg, wu);
  gather_wd_kernel<<<dim3(HIDDEN, (KPAD + 2047) / 2048), blk, 0, stream>>>(Wd, id, wd);

  gateup_kernel<<<dim3(TOKENS / TM, KPAD / TN),   blk, 0, stream>>>(xb, wg, wu, hb);
  down_kernel  <<<dim3(TOKENS / TM, HIDDEN / TN), blk, 0, stream>>>(hb, wd, out);
}
